// BasicBlock_75917841924279
// MI455X (gfx1250) — matrix operand order repaired and checked by reading, not compiled, not run on hardware
//
#include <hip/hip_runtime.h>

#define BB 4
#define CCH 64
#define HH 160
#define WW 160
#define HWP (HH * WW)
#define PIX (BB * HWP)
#define OMS 32   // padded channel stride for offset/mask tensors (27 used)

typedef __attribute__((ext_vector_type(16))) __bf16 v16bf;
typedef __attribute__((ext_vector_type(8)))  float  v8f;

__device__ __forceinline__ unsigned short f2bf(float f) {
    union { float f; unsigned int u; } cv; cv.f = f;
    unsigned int u = cv.u + 0x7FFFu + ((cv.u >> 16) & 1u);  // round-to-nearest-even
    return (unsigned short)(u >> 16);
}

// ---------------------------------------------------------------------------
// NCHW f32 -> NHWC f32 (channel-contiguous so bilinear gathers are coalesced)
// ---------------------------------------------------------------------------
__global__ void nchw_to_nhwc(const float* __restrict__ in, float* __restrict__ out) {
    int id = blockIdx.x * 256 + threadIdx.x;
    if (id >= PIX * CCH) return;
    int c = id & 63;
    int p = id >> 6;
    int b = p / HWP;
    int yx = p - b * HWP;
    out[id] = in[((size_t)(b * CCH + c)) * HWP + yx];
}

// ---------------------------------------------------------------------------
// Pack OIHW f32 weights -> bf16 WMMA B fragments (32x16 per kstep) in this code's contiguous-k
// layout: element (k,n): lane = (n&15) + 16*(k>>4), idx = k&15.
// K ordering: k_global = tap*64 + c  (tap = 3x3 position, c = in-channel)
// out[nt][kstep][lane][idx], kstep = 0..17
// ---------------------------------------------------------------------------
__global__ void pack_weights(const float* __restrict__ w, unsigned short* __restrict__ out,
                             int Cout, int NT) {
    int id = blockIdx.x * 256 + threadIdx.x;
    if (id >= NT * 18 * 32 * 16) return;
    int i     = id & 15;
    int lane  = (id >> 4) & 31;
    int kstep = (id >> 9) % 18;
    int nt    = (id >> 9) / 18;
    int n = nt * 16 + (lane & 15);
    int k = kstep * 32 + ((lane >> 4) << 4) + i;
    int tap = k / 64;
    int c   = k - tap * 64;
    float v = (n < Cout) ? w[((size_t)n * CCH + c) * 9 + tap] : 0.0f;
    out[id] = f2bf(v);
}

// ---------------------------------------------------------------------------
// Generic (modulated-deformable) 3x3 conv over 64 input channels as a WMMA
// GEMM: M = 16 pixels per wave, N = NT*16 out channels, K = 576 (9 taps * 64).
// All variant flags are compile-time so the hot loop carries no branches.
// ---------------------------------------------------------------------------
template <int NT, bool DEFORM, bool DO_BN, bool RESID, bool GELU, bool NCHW_OUT>
__global__ void __launch_bounds__(256) dconv_wmma(
    const float* __restrict__ in_nhwc,
    const float* __restrict__ om,            // [PIX, 32]: 0..17 offsets, 18..26 mask logits
    const unsigned short* __restrict__ wpack,// packed bf16 B fragments
    const float* __restrict__ bias,
    const float* __restrict__ bn_g, const float* __restrict__ bn_b,
    const float* __restrict__ bn_m, const float* __restrict__ bn_v,
    const float* __restrict__ residual,      // NHWC
    float* __restrict__ out,
    int Cout, int out_stride) {

    __shared__ __attribute__((aligned(32))) unsigned short lds[8][2][32][16];

    int lane = threadIdx.x & 31;
    int wave = threadIdx.x >> 5;
    int tile = blockIdx.x * 8 + wave;
    int m0 = tile * 16;
    if (m0 >= PIX) return;   // wave-uniform: EXEC stays all-ones for WMMA

    // Warm the cache hierarchy for data this tile will touch (global_prefetch_b8).
    {
        int pb = m0 / HWP;
        int prem = m0 - pb * HWP;
        int pyr = prem / WW;
        int y0r = (pyr > 0) ? (pyr - 1) : 0;
        const float* pf = in_nhwc + ((size_t)pb * HWP + (size_t)y0r * WW) * CCH;
        __builtin_prefetch(pf + (size_t)lane * 64, 0, 3);
        if (DEFORM) __builtin_prefetch(om + (size_t)(m0 + (lane & 15)) * OMS, 0, 3);
    }

    v8f acc[NT] = {};

    // This lane produces channels c0, c0+1 of every sample.
    int c0 = lane * 2;
    int ks_w    = c0 >> 5;                                       // which 32-wide K step
    int kl      = c0 & 31;                                       // K within step
    int idxw    = kl & 15;                                       // A-frag elem
    int lanebit = kl >> 4;                                       // A-frag lane half

    for (int tap = 0; tap < 9; ++tap) {
        int ti = tap / 3, tj = tap % 3;

        // Hoist B-fragment loads: their latency hides behind the sampling VALU work.
        v16bf bfrag[2][NT];
#pragma unroll
        for (int ks = 0; ks < 2; ++ks) {
            int kstep = tap * 2 + ks;
#pragma unroll
            for (int nt = 0; nt < NT; ++nt) {
                bfrag[ks][nt] = *(const v16bf*)(wpack +
                    ((((size_t)nt * 18 + kstep) * 32 + lane) << 4));
            }
        }

        for (int mi = 0; mi < 16; ++mi) {
            int p = m0 + mi;
            int b = p / HWP;
            int rem = p - b * HWP;
            int y = rem / WW;
            int x = rem - y * WW;
            float py, px, msk;
            if (DEFORM) {
                const float* op = om + (size_t)p * OMS;
                float dy = op[2 * tap];
                float dx = op[2 * tap + 1];
                msk = 1.0f / (1.0f + __expf(-op[18 + tap]));
                py = (float)(y - 1 + ti) + dy;
                px = (float)(x - 1 + tj) + dx;
            } else {
                py = (float)(y - 1 + ti);
                px = (float)(x - 1 + tj);
                msk = 1.0f;
            }
            float y0f = floorf(py), x0f = floorf(px);
            float ly = py - y0f, lx = px - x0f;
            int y0 = (int)y0f, x0 = (int)x0f;
            float v0 = 0.0f, v1 = 0.0f;
            const float* base = in_nhwc + (size_t)b * HWP * CCH + c0;
#pragma unroll
            for (int cn = 0; cn < 4; ++cn) {
                int yy = y0 + (cn >> 1);
                int xx = x0 + (cn & 1);
                float wy = (cn >> 1) ? ly : (1.0f - ly);
                float wx = (cn & 1) ? lx : (1.0f - lx);
                float wgt = wy * wx;
                if (wgt != 0.0f && yy >= 0 && yy < HH && xx >= 0 && xx < WW) {
                    const float* sp = base + ((size_t)yy * WW + xx) * CCH;
                    v0 += wgt * sp[0];
                    v1 += wgt * sp[1];
                }
            }
            v0 *= msk;
            v1 *= msk;
            int lw = mi + (lanebit << 4);
            lds[wave][ks_w][lw][idxw]     = f2bf(v0);
            lds[wave][ks_w][lw][idxw + 1] = f2bf(v1);
        }
        // wave-local LDS producer->consumer: in-order DS, just drain the counter
        asm volatile("s_wait_dscnt 0" ::: "memory");
#pragma unroll
        for (int ks = 0; ks < 2; ++ks) {
            v16bf afrag = *(const v16bf*)(&lds[wave][ks][lane][0]);
#pragma unroll
            for (int nt = 0; nt < NT; ++nt) {
                acc[nt] = __builtin_amdgcn_wmma_f32_16x16x32_bf16(
                    false, afrag, false, bfrag[ks][nt], (short)0, acc[nt], false, false);
            }
        }
        asm volatile("s_wait_dscnt 0" ::: "memory");  // A reads done before next-tap stores
    }

    // Epilogue. D layout: VGPR r, lanes 0-15 -> M=r,N=lane; lanes 16-31 -> M=r+8.
    int nl = lane & 15;
    int mh = (lane >> 4) * 8;
#pragma unroll
    for (int nt = 0; nt < NT; ++nt) {
        int n = nt * 16 + nl;
        if (n < Cout) {
            float bs = (bias != nullptr) ? bias[n] : 0.0f;
            float g = 1.0f, bb2 = 0.0f, mn = 0.0f, iv = 1.0f;
            if (DO_BN) {
                g = bn_g[n]; bb2 = bn_b[n]; mn = bn_m[n];
                iv = rsqrtf(bn_v[n] + 1e-5f);
            }
#pragma unroll
            for (int r = 0; r < 8; ++r) {
                int p = m0 + mh + r;
                float v = acc[nt][r] + bs;
                if (DO_BN) v = (v - mn) * iv * g + bb2;
                if (RESID) v += residual[(size_t)p * CCH + n];
                if (GELU) v = 0.5f * v * (1.0f + erff(v * 0.70710678118f));
                if (NCHW_OUT) {
                    int b = p / HWP;
                    int rem = p - b * HWP;
                    out[((size_t)(b * Cout + n)) * HWP + rem] = v;
                } else {
                    out[(size_t)p * out_stride + n] = v;
                }
            }
        }
    }
}

// ---------------------------------------------------------------------------
extern "C" void kernel_launch(void* const* d_in, const int* in_sizes, int n_in,
                              void* d_out, int out_size, void* d_ws, size_t ws_size,
                              hipStream_t stream) {
    (void)in_sizes; (void)n_in; (void)out_size; (void)ws_size;

    const float* x      = (const float*)d_in[0];
    const float* w_om1  = (const float*)d_in[1];
    const float* b_om1  = (const float*)d_in[2];
    const float* w_dcn1 = (const float*)d_in[3];
    const float* bn1_g  = (const float*)d_in[4];
    const float* bn1_b  = (const float*)d_in[5];
    const float* bn1_m  = (const float*)d_in[6];
    const float* bn1_v  = (const float*)d_in[7];
    const float* w_om2  = (const float*)d_in[8];
    const float* b_om2  = (const float*)d_in[9];
    const float* w_dcn2 = (const float*)d_in[10];
    const float* bn2_g  = (const float*)d_in[11];
    const float* bn2_b  = (const float*)d_in[12];
    const float* bn2_m  = (const float*)d_in[13];
    const float* bn2_v  = (const float*)d_in[14];

    float* ws = (float*)d_ws;
    float* x_nhwc = ws;                              // PIX*64
    float* y1     = x_nhwc + (size_t)PIX * 64;       // PIX*64
    float* om1    = y1     + (size_t)PIX * 64;       // PIX*32
    float* om2    = om1    + (size_t)PIX * 32;       // PIX*32
    unsigned short* wp_om1  = (unsigned short*)(om2 + (size_t)PIX * 32);
    unsigned short* wp_dcn1 = wp_om1  + 2 * 18 * 32 * 16;
    unsigned short* wp_om2  = wp_dcn1 + 4 * 18 * 32 * 16;
    unsigned short* wp_dcn2 = wp_om2  + 2 * 18 * 32 * 16;

    nchw_to_nhwc<<<(PIX * CCH + 255) / 256, 256, 0, stream>>>(x, x_nhwc);

    pack_weights<<<(2 * 9216 + 255) / 256, 256, 0, stream>>>(w_om1,  wp_om1,  27, 2);
    pack_weights<<<(4 * 9216 + 255) / 256, 256, 0, stream>>>(w_dcn1, wp_dcn1, 64, 4);
    pack_weights<<<(2 * 9216 + 255) / 256, 256, 0, stream>>>(w_om2,  wp_om2,  27, 2);
    pack_weights<<<(4 * 9216 + 255) / 256, 256, 0, stream>>>(w_dcn2, wp_dcn2, 64, 4);

    const int nblocks = PIX / 16 / 8;  // 6400 M-tiles, 8 waves/block -> 800

    // 1) offset/mask conv #1 (plain conv, bias, raw logits stored)
    dconv_wmma<2, false, false, false, false, false><<<nblocks, 256, 0, stream>>>(
        x_nhwc, nullptr, wp_om1, b_om1,
        nullptr, nullptr, nullptr, nullptr, nullptr,
        om1, 27, OMS);

    // 2) DCN #1 + BN1 + GELU -> y1 (NHWC)
    dconv_wmma<4, true, true, false, true, false><<<nblocks, 256, 0, stream>>>(
        x_nhwc, om1, wp_dcn1, nullptr,
        bn1_g, bn1_b, bn1_m, bn1_v, nullptr,
        y1, 64, 64);

    // 3) offset/mask conv #2
    dconv_wmma<2, false, false, false, false, false><<<nblocks, 256, 0, stream>>>(
        y1, nullptr, wp_om2, b_om2,
        nullptr, nullptr, nullptr, nullptr, nullptr,
        om2, 27, OMS);

    // 4) DCN #2 + BN2 + residual(x) + GELU -> d_out (NCHW)
    dconv_wmma<4, true, true, true, true, true><<<nblocks, 256, 0, stream>>>(
        y1, om2, wp_dcn2, nullptr,
        bn2_g, bn2_b, bn2_m, bn2_v, x_nhwc,
        (float*)d_out, 64, 64);
}
